// DeformNet_18889266168040
// MI455X (gfx1250) — compile-verified, matrix operand order checked by reading, not run on hardware
//
#include <hip/hip_runtime.h>
#include <hip/hip_bf16.h>
#include <math.h>

// ---------------- problem constants ----------------
#define BB   4
#define HH   448
#define WW   640
#define HQ   112      // H/4
#define WQ   160      // W/4
#define HHF  224      // H/2
#define WHF  320      // W/2
#define CF   565      // features2 channels
#define CFP  576      // pad 565 -> multiple of 32 (NHWC channel pad)
#define K1P  2304     // 4*576
#define CD1  32       // deconv1 out channels
#define K2P  128      // 4*32
#define CD2  16       // deconv2 out channels
#define C0P  32       // 28 -> 32 channel pad for c0 input
#define K0P  288      // 9*32
#define CR   16       // res-block channels
#define KRP  160      // pad(9*16 = 144) to 32
#define KMAXC 288     // max Kp for conv3 stack
#define BTS  296      // Bt row stride (Kp max + 8 pad, keeps 16B row alignment)

typedef __attribute__((ext_vector_type(16))) _Float16 v16h;
typedef __attribute__((ext_vector_type(8)))  float    v8f;
typedef unsigned int u32x4 __attribute__((ext_vector_type(4)));
typedef int          i32x8 __attribute__((ext_vector_type(8)));
typedef int          i32x4 __attribute__((ext_vector_type(4)));

#if defined(__has_builtin)
# if __has_builtin(__builtin_amdgcn_tensor_load_to_lds) && __has_builtin(__builtin_amdgcn_s_wait_tensorcnt)
#  define HAVE_TDM 1
# else
#  define HAVE_TDM 0
# endif
#else
# define HAVE_TDM 0
#endif

__device__ __forceinline__ int imin(int a, int b) { return a < b ? a : b; }
__device__ __forceinline__ int imax(int a, int b) { return a > b ? a : b; }

#if HAVE_TDM
// 1D contiguous TDM load: nelem f16 elements from gaddr -> LDS[lds_addr].
__device__ __forceinline__ void tdm_load_1d(unsigned lds_addr, const void* gaddr, unsigned nelem) {
    unsigned long long ga = (unsigned long long)(uintptr_t)gaddr;
    u32x4 g0;
    g0.x = 1u;                                              // count=1 valid user descriptor
    g0.y = lds_addr;                                        // lds_addr [63:32]
    g0.z = (unsigned)(ga & 0xffffffffu);                    // global_addr low
    g0.w = (unsigned)((ga >> 32) & 0x01ffffffu) | 0x80000000u; // addr[56:32] | type=2
    i32x8 g1;
    g1[0] = 0x00010000;                                     // data_size = 2B, no multicast
    g1[1] = (int)((nelem & 0xffffu) << 16);                 // tensor_dim0[15:0] at bits 63:48
    g1[2] = (int)((nelem >> 16) | (1u << 16));              // tensor_dim0[31:16] | tensor_dim1=1
    g1[3] = (int)((nelem & 0xffffu) << 16);                 // tile_dim0 = nelem (bits 127:112)
    g1[4] = 1;                                              // tile_dim1 = 1
    g1[5] = (int)nelem;                                     // tensor_dim0_stride
    g1[6] = 0;
    g1[7] = 0;
    i32x4 z4 = {0, 0, 0, 0};
#if __clang_major__ >= 23
    i32x8 z8 = {0, 0, 0, 0, 0, 0, 0, 0};
    __builtin_amdgcn_tensor_load_to_lds(g0, g1, z4, z4, z8, 0);
#else
    __builtin_amdgcn_tensor_load_to_lds(g0, g1, z4, z4, 0);
#endif
}
#endif

// ---------------- features: f32 NCHW -> f16 NHWC (channel-padded to 576) ----------------
__global__ void k_feat_nhwc(const float* __restrict__ src, _Float16* __restrict__ dst) {
    const long N = (long)BB * HQ * WQ * CFP;
    long i = (long)blockIdx.x * blockDim.x + threadIdx.x;
    if (i >= N) return;
    int c = (int)(i % CFP);
    long p = i / CFP;
    int pix = (int)(p % (HQ * WQ));
    int b = (int)(p / (HQ * WQ));
    _Float16 v = (_Float16)0.f;
    if (c < CF) v = (_Float16)src[((long)b * CF + c) * (HQ * WQ) + pix];
    dst[i] = v;
}

// ---------------- front-end: flow resize + warp + grid_sample + validity ----------------
// Writes: d_out[2*BHW .. 5*BHW) = target_matches (B,3,H,W)
//         d_out[5*BHW .. 6*BHW) = valid_correspondences (B,H,W)
//         c0in (NHWC, 32ch) channels 16..31 = mask_input (12ch) + zero pad
__global__ void k_frontend(const float* __restrict__ src, const float* __restrict__ tgt,
                           const int* __restrict__ anchors, const float* __restrict__ flow2,
                           _Float16* __restrict__ c0in, float* __restrict__ out) {
    const long BHW = (long)BB * HH * WW;
    long idx = (long)blockIdx.x * blockDim.x + threadIdx.x;
    if (idx >= BHW) return;
    int x = (int)(idx % WW);
    long t = idx / WW;
    int y = (int)(t % HH);
    int b = (int)(t / HH);
    const long HW = (long)HH * WW;

    // resize_bilinear of flow2 (scale 0.25), x20
    float sy = fminf(fmaxf((y + 0.5f) * 0.25f - 0.5f, 0.f), (float)(HQ - 1));
    float sx = fminf(fmaxf((x + 0.5f) * 0.25f - 0.5f, 0.f), (float)(WQ - 1));
    int fy0 = (int)floorf(sy), fx0 = (int)floorf(sx);
    int fy1 = imin(fy0 + 1, HQ - 1), fx1 = imin(fx0 + 1, WQ - 1);
    float tyf = sy - fy0, txf = sx - fx0;
    const float* f2 = flow2 + (long)b * 2 * HQ * WQ;
    float fl[2];
    #pragma unroll
    for (int c = 0; c < 2; ++c) {
        const float* p = f2 + (long)c * HQ * WQ;
        float v00 = p[fy0 * WQ + fx0], v01 = p[fy0 * WQ + fx1];
        float v10 = p[fy1 * WQ + fx0], v11 = p[fy1 * WQ + fx1];
        fl[c] = 20.f * ((v00 * (1.f - txf) + v01 * txf) * (1.f - tyf) +
                        (v10 * (1.f - txf) + v11 * txf) * tyf);
    }

    float xw = (float)x + fl[0], yw = (float)y + fl[1];
    float gxn = xw * (2.f / (WW - 1)) - 1.f;
    float gyn = yw * (2.f / (HH - 1)) - 1.f;
    float gx = ((gxn + 1.f) * WW - 1.f) * 0.5f;
    float gy = ((gyn + 1.f) * HH - 1.f) * 0.5f;
    int ix0 = (int)floorf(gx), iy0 = (int)floorf(gy);
    int ix1 = ix0 + 1, iy1 = iy0 + 1;
    float wx = gx - ix0, wy = gy - iy0;
    float m00 = (ix0 >= 0 && ix0 < WW && iy0 >= 0 && iy0 < HH) ? 1.f : 0.f;
    float m01 = (ix1 >= 0 && ix1 < WW && iy0 >= 0 && iy0 < HH) ? 1.f : 0.f;
    float m10 = (ix0 >= 0 && ix0 < WW && iy1 >= 0 && iy1 < HH) ? 1.f : 0.f;
    float m11 = (ix1 >= 0 && ix1 < WW && iy1 >= 0 && iy1 < HH) ? 1.f : 0.f;
    int cx0 = imin(imax(ix0, 0), WW - 1), cx1 = imin(imax(ix1, 0), WW - 1);
    int cy0 = imin(imax(iy0, 0), HH - 1), cy1 = imin(imax(iy1, 0), HH - 1);
    long o00 = (long)cy0 * WW + cx0, o01 = (long)cy0 * WW + cx1;
    long o10 = (long)cy1 * WW + cx0, o11 = (long)cy1 * WW + cx1;
    float w00 = (1.f - wx) * (1.f - wy), w01 = wx * (1.f - wy);
    float w10 = (1.f - wx) * wy,         w11 = wx * wy;

    const float* tb = tgt + (long)b * 6 * HW;
    float rgbw[3], match[3];
    #pragma unroll
    for (int c = 0; c < 3; ++c) {
        const float* p = tb + (long)c * HW;
        rgbw[c] = p[o00]*m00*w00 + p[o01]*m01*w01 + p[o10]*m10*w10 + p[o11]*m11*w11;
        const float* q = tb + (long)(3 + c) * HW;
        match[c] = q[o00]*m00*w00 + q[o01]*m01*w01 + q[o10]*m10*w10 + q[o11]*m11*w11;
    }
    const float* qz = tb + (long)5 * HW;
    float vz00 = (qz[o00] > 0.f && qz[o00] <= 6.f) ? 1.f : 0.f;
    float vz01 = (qz[o01] > 0.f && qz[o01] <= 6.f) ? 1.f : 0.f;
    float vz10 = (qz[o10] > 0.f && qz[o10] <= 6.f) ? 1.f : 0.f;
    float vz11 = (qz[o11] > 0.f && qz[o11] <= 6.f) ? 1.f : 0.f;
    float tmv = vz00*m00*w00 + vz01*m01*w01 + vz10*m10*w10 + vz11*m11*w11;
    bool tm_valid = tmv >= 0.999f;

    const float* sb = src + (long)b * 6 * HW;
    float sz = sb[5 * HW + (long)y * WW + x];
    const int* an = anchors + ((long)b * HH + y) * (long)WW * 4 + (long)x * 4;
    bool anchor_valid = (an[0] >= 0) && (an[1] >= 0) && (an[2] >= 0) && (an[3] >= 0);
    bool valid_src = (sz > 0.f) && (sz <= 6.f) && anchor_valid;
    bool valid_tgt = (match[2] > 0.f) && (match[2] <= 6.f) && tm_valid;
    float validf = (valid_src && valid_tgt) ? 1.f : 0.f;

    #pragma unroll
    for (int c = 0; c < 3; ++c)
        out[2 * BHW + ((long)(b * 3 + c) * HH + y) * WW + x] = match[c];
    out[5 * BHW + idx] = validf;

    // mask_input into c0in NHWC channels 16..31 (contiguous per pixel)
    _Float16* cb = c0in + (((long)b * HH + y) * WW + x) * C0P;
    #pragma unroll
    for (int c = 0; c < 6; ++c)
        cb[16 + c] = (_Float16)sb[(long)c * HW + (long)y * WW + x];
    #pragma unroll
    for (int c = 0; c < 3; ++c) cb[22 + c] = (_Float16)rgbw[c];
    #pragma unroll
    for (int c = 0; c < 3; ++c) cb[25 + c] = (_Float16)match[c];
    #pragma unroll
    for (int c = 28; c < 32; ++c) cb[c] = (_Float16)0.f;
}

// ---------------- weight repack ----------------
// deconv OIHW (O,Cin,4,4) -> wp[parity][Kp][Cout], k = tap*Cpad + c
__global__ void k_pack_deconv(const float* __restrict__ w, _Float16* __restrict__ wp,
                              int Cin, int Cpad, int Cout, int Kp) {
    int idx = blockIdx.x * blockDim.x + threadIdx.x;
    int total = 4 * Kp * Cout;
    if (idx >= total) return;
    int o = idx % Cout;
    int r = idx / Cout;
    int k = r % Kp;
    int p = r / Kp;
    int py = p >> 1, px = p & 1;
    int tap = k / Cpad, c = k - tap * Cpad;
    _Float16 v = (_Float16)0.f;
    if (c < Cin) {
        int ky = 2 * (tap >> 1) + py;
        int kx = 2 * (tap & 1) + px;
        v = (_Float16)w[(((long)o * Cin + c) * 4 + ky) * 4 + kx];
    }
    wp[idx] = v;
}

// conv3x3 OIHW (O,Cin,3,3) -> wc[Kp][16], k = tap*Cpad + c
__global__ void k_pack_conv3(const float* __restrict__ w, _Float16* __restrict__ wc,
                             int Cin, int Cpad, int Cout, int Kp) {
    int idx = blockIdx.x * blockDim.x + threadIdx.x;
    int total = Kp * 16;
    if (idx >= total) return;
    int n = idx % 16;
    int k = idx / 16;
    int tap = k / Cpad, c = k % Cpad;
    _Float16 v = (_Float16)0.f;
    if (tap < 9 && c < Cin && n < Cout) {
        int ky = tap / 3, kx = tap % 3;
        v = (_Float16)w[(((long)n * Cin + c) * 3 + ky) * 3 + kx];
    }
    wc[idx] = v;
}

// ---------------- WMMA deconv (lhs-dilated conv, per-parity GEMM), NHWC ----------------
// C tile = 64 output pixels x 16 channels; 4 waves/block; A+B double-buffered in LDS.
__global__ void __launch_bounds__(128)
k_deconv_wmma(const _Float16* __restrict__ in, int Cpad, int Hi, int Wi,
              const _Float16* __restrict__ wp, int Kp, int Ncols,
              const float* __restrict__ bias,
              _Float16* __restrict__ out, int outChanBase, int outCTot) {
    __shared__ _Float16 As[2][64][40];   // 40-half row stride: 80B, 16B-aligned
    __shared__ _Float16 Bs[2][16][40];   // transposed [n][k]
    int bp = blockIdx.z;
    int b = bp >> 2, p = bp & 3, py = p >> 1, px = p & 1;
    int ntile = blockIdx.y;
    int row0 = blockIdx.x * 64;
    int tid = threadIdx.x, lane = tid & 31, wave = tid >> 5;
    int Ho = Hi * 2, Wo = Wi * 2;
    int ar = tid >> 1, ah = tid & 1;
    int g = row0 + ar;
    int oyh = g / Wi, oxh = g - oyh * Wi;

    auto stageA = [&](int kc, int buf) {
        int kbase = (kc << 5) + (ah << 4);     // multiple of 16; Cpad multiple of 16
        int tap = kbase / Cpad;                // group never crosses a tap boundary
        int c0 = kbase - tap * Cpad;
        int iy = oyh + (tap >> 1) + py - 1;
        int ix = oxh + (tap & 1) + px - 1;
        _Float16* dp = &As[buf][ar][ah << 4];
        if ((unsigned)iy < (unsigned)Hi && (unsigned)ix < (unsigned)Wi) {
            const _Float16* sp = in + (((long)b * Hi + iy) * Wi + ix) * Cpad + c0;
            #pragma unroll
            for (int i = 0; i < 16; ++i) dp[i] = sp[i];
        } else {
            #pragma unroll
            for (int i = 0; i < 16; ++i) dp[i] = (_Float16)0.f;
        }
    };
    auto stageB = [&](int kc, int buf) {
        int kk = tid >> 2, n4 = (tid & 3) << 2;
        const _Float16* ws = wp + ((long)p * Kp + (kc << 5) + kk) * Ncols + ntile * 16 + n4;
        #pragma unroll
        for (int j = 0; j < 4; ++j) Bs[buf][n4 + j][kk] = ws[j];
    };

    v8f acc = {};
    int nchunks = Kp >> 5;
    stageA(0, 0);
    stageB(0, 0);
    __syncthreads();
    for (int kc = 0; kc < nchunks; ++kc) {
        int cur = kc & 1;
        if (kc + 1 < nchunks) {
            __builtin_prefetch(wp + ((long)p * Kp + ((kc + 2) << 5)) * Ncols, 0, 1);
            stageA(kc + 1, cur ^ 1);
            stageB(kc + 1, cur ^ 1);
        }
        v16h afr, bfr;
        int mr = (wave << 4) + (lane & 15);
        int kh = (lane >> 4) << 4;
        #pragma unroll
        for (int i = 0; i < 16; ++i) afr[i] = As[cur][mr][kh + i];
        #pragma unroll
        for (int i = 0; i < 16; ++i) bfr[i] = Bs[cur][lane & 15][kh + i];
        acc = __builtin_amdgcn_wmma_f32_16x16x32_f16(false, afr, false, bfr,
                                                     (short)0, acc, false, false);
        __syncthreads();
    }
    // epilogue: bias, NHWC scatter to interleaved (x2 upsample) positions
    int n = lane & 15;
    int ch = ntile * 16 + n;
    float bv = bias[ch];
    int mbase = (wave << 4) + ((lane >> 4) << 3);
    #pragma unroll
    for (int j = 0; j < 8; ++j) {
        int gg = row0 + mbase + j;
        int oy2 = gg / Wi, ox2 = gg - oy2 * Wi;
        int oy = oy2 * 2 + py, ox = ox2 * 2 + px;
        float v = acc[j] + bv;
        out[(((long)b * Ho + oy) * Wo + ox) * outCTot + outChanBase + ch] = (_Float16)v;
    }
}

// ---------------- WMMA conv3x3 (fused im2col GEMM), NHWC, N = 16 ----------------
// Weights persist in LDS (loaded once, via TDM when available); A double-buffered.
#define EP_RELU     1
#define EP_RESID    2
#define EP_SIGMOID  3
__global__ void __launch_bounds__(128)
k_conv3_wmma(const _Float16* __restrict__ in, int cshift, int Kp,
             const _Float16* __restrict__ wc, const float* __restrict__ bias, int Cout,
             const _Float16* __restrict__ resid, _Float16* __restrict__ dst, int dstC,
             float* __restrict__ dstf, int mode) {
    __shared__ _Float16 As[2][64][40];
    __shared__ _Float16 Bt[16][BTS];        // transposed weights [n][k]
    __shared__ _Float16 braw[KMAXC * 16];   // raw staging for TDM / transpose source
    int row0 = blockIdx.x * 64;
    int tid = threadIdx.x, lane = tid & 31, wave = tid >> 5;
    int ar = tid >> 1, ah = tid & 1;
    int g = row0 + ar;
    int x = g % WW;
    int t2 = g / WW;
    int y = t2 % HH;
    int b = t2 / HH;
    int Cpad = 1 << cshift, cm = Cpad - 1;
    const _Float16* inb = in + ((long)b * HH * WW << cshift);

    // --- load weights into LDS once ---
#if HAVE_TDM
    if (wave == 0) {
        tdm_load_1d((unsigned)(uintptr_t)&braw[0], wc, (unsigned)(Kp * 16));
        __builtin_amdgcn_s_wait_tensorcnt(0);
    }
    __syncthreads();
    for (int kk = tid; kk < Kp; kk += 128) {
        #pragma unroll
        for (int n = 0; n < 16; ++n) Bt[n][kk] = braw[kk * 16 + n];
    }
#else
    for (int kk = tid; kk < Kp; kk += 128) {
        #pragma unroll
        for (int n = 0; n < 16; ++n) Bt[n][kk] = wc[kk * 16 + n];
    }
#endif

    auto stageA = [&](int kc, int buf) {
        int kbase = (kc << 5) + (ah << 4);   // multiple of 16; never crosses a tap
        int tap = kbase >> cshift;
        int c0 = kbase & cm;
        _Float16* dp = &As[buf][ar][ah << 4];
        if (tap < 9) {
            int dy = tap / 3 - 1, dx = tap % 3 - 1;
            int yy = y + dy, xx = x + dx;
            if ((unsigned)yy < (unsigned)HH && (unsigned)xx < (unsigned)WW) {
                const _Float16* sp = inb + (((long)yy * WW + xx) << cshift) + c0;
                #pragma unroll
                for (int i = 0; i < 16; ++i) dp[i] = sp[i];
                return;
            }
        }
        #pragma unroll
        for (int i = 0; i < 16; ++i) dp[i] = (_Float16)0.f;
    };

    v8f acc = {};
    int nchunks = Kp >> 5;
    stageA(0, 0);
    __syncthreads();
    for (int kc = 0; kc < nchunks; ++kc) {
        int cur = kc & 1;
        if (kc + 1 < nchunks) stageA(kc + 1, cur ^ 1);
        v16h afr, bfr;
        int mr = (wave << 4) + (lane & 15);
        int kh = (lane >> 4) << 4;
        #pragma unroll
        for (int i = 0; i < 16; ++i) afr[i] = As[cur][mr][kh + i];
        #pragma unroll
        for (int i = 0; i < 16; ++i) bfr[i] = Bt[lane & 15][(kc << 5) + kh + i];
        acc = __builtin_amdgcn_wmma_f32_16x16x32_f16(false, afr, false, bfr,
                                                     (short)0, acc, false, false);
        __syncthreads();
    }

    int n = lane & 15;
    float bv = (n < Cout) ? bias[n] : 0.f;
    int mbase = (wave << 4) + ((lane >> 4) << 3);
    #pragma unroll
    for (int j = 0; j < 8; ++j) {
        int gg = row0 + mbase + j;
        int xx = gg % WW;
        int tt = gg / WW;
        int yy = tt % HH;
        int b2 = tt / HH;
        long pix = ((long)b2 * HH + yy) * WW + xx;
        float v = acc[j] + bv;
        if (mode == EP_SIGMOID) {
            if (n == 0) {
                float s = 1.f / (1.f + expf(-v));
                dstf[pix] = s;                               // mask_pred
                dstf[pix + (long)BB * HH * WW] = s;          // correspondence_weights
            }
        } else {
            if (mode == EP_RELU) {
                v = fmaxf(v, 0.f);
            } else if (mode == EP_RESID) {
                float r = (float)resid[pix * dstC + n];
                v = fmaxf(v + r, 0.f);
            }
            if (n < Cout) dst[pix * dstC + n] = (_Float16)v;
        }
    }
}

// ---------------- host orchestration ----------------
extern "C" void kernel_launch(void* const* d_in, const int* in_sizes, int n_in,
                              void* d_out, int out_size, void* d_ws, size_t ws_size,
                              hipStream_t stream) {
    (void)in_sizes; (void)n_in; (void)out_size; (void)ws_size;
    const float* src      = (const float*)d_in[0];
    const float* tgt      = (const float*)d_in[1];
    const int*   anchors  = (const int*)  d_in[6];
    const float* flow2    = (const float*)d_in[10];
    const float* features = (const float*)d_in[11];
    const float* uc1_w = (const float*)d_in[12];
    const float* uc1_b = (const float*)d_in[13];
    const float* uc2_w = (const float*)d_in[14];
    const float* uc2_b = (const float*)d_in[15];
    const float* c0_w  = (const float*)d_in[16];
    const float* c0_b  = (const float*)d_in[17];
    const float* f_w   = (const float*)d_in[18];
    const float* f_b   = (const float*)d_in[19];
    const float* rw_[6]; const float* rb_[6];
    for (int i = 0; i < 6; ++i) {
        rw_[i] = (const float*)d_in[20 + 2 * i];
        rb_[i] = (const float*)d_in[21 + 2 * i];
    }
    float* out = (float*)d_out;

    char* ws = (char*)d_ws;
    size_t off = 0;
    auto carve = [&](size_t bytes) -> void* {
        void* p = ws + off;
        off = (off + bytes + 255) & ~(size_t)255;
        return p;
    };
    const long HW = (long)HH * WW;
    const long NFE = (long)BB * HQ * WQ * CFP;
    _Float16* feat16 = (_Float16*)carve((size_t)NFE * 2);                     // NHWC, 576ch
    _Float16* w1p    = (_Float16*)carve((size_t)4 * K1P * CD1 * 2);
    _Float16* d1out  = (_Float16*)carve((size_t)BB * HHF * WHF * CD1 * 2);    // NHWC, 32ch
    _Float16* w2p    = (_Float16*)carve((size_t)4 * K2P * CD2 * 2);
    _Float16* c0in   = (_Float16*)carve((size_t)BB * HW * C0P * 2);           // NHWC, 32ch
    _Float16* c0wp   = (_Float16*)carve((size_t)K0P * 16 * 2);
    _Float16* rwp[6];
    for (int i = 0; i < 6; ++i) rwp[i] = (_Float16*)carve((size_t)KRP * 16 * 2);
    _Float16* fwp    = (_Float16*)carve((size_t)KRP * 16 * 2);
    _Float16* h1     = (_Float16*)carve((size_t)BB * HW * CR * 2);            // NHWC, 16ch
    _Float16* h2     = (_Float16*)carve((size_t)BB * HW * CR * 2);

    const long BHW = (long)BB * HH * WW;

    // 1) features2 -> f16 NHWC (padded channels)
    k_feat_nhwc<<<(int)((NFE + 255) / 256), 256, 0, stream>>>(features, feat16);

    // 2) front-end
    k_frontend<<<(int)((BHW + 255) / 256), 256, 0, stream>>>(src, tgt, anchors, flow2, c0in, out);

    // 3) weight repacks
    {
        int tot = 4 * K1P * CD1;
        k_pack_deconv<<<(tot + 255) / 256, 256, 0, stream>>>(uc1_w, w1p, CF, CFP, CD1, K1P);
    }
    {
        int tot = 4 * K2P * CD2;
        k_pack_deconv<<<(tot + 255) / 256, 256, 0, stream>>>(uc2_w, w2p, CD1, CD1, CD2, K2P);
    }
    k_pack_conv3<<<(K0P * 16 + 255) / 256, 256, 0, stream>>>(c0_w, c0wp, 28, C0P, 16, K0P);
    for (int i = 0; i < 6; ++i)
        k_pack_conv3<<<(KRP * 16 + 255) / 256, 256, 0, stream>>>(rw_[i], rwp[i], CR, CR, 16, KRP);
    k_pack_conv3<<<(KRP * 16 + 255) / 256, 256, 0, stream>>>(f_w, fwp, CR, CR, 1, KRP);

    // 4) deconv1: 565(576) -> 32, 112x160 -> 224x320   (M/class = 17920 = 280*64)
    k_deconv_wmma<<<dim3(280, CD1 / 16, BB * 4), 128, 0, stream>>>(
        feat16, CFP, HQ, WQ, w1p, K1P, CD1, uc1_b, d1out, 0, CD1);

    // 5) deconv2: 32 -> 16, 224x320 -> 448x640, into c0in channels 0..15 (M/class = 71680)
    k_deconv_wmma<<<dim3(1120, CD2 / 16, BB * 4), 128, 0, stream>>>(
        d1out, CD1, HHF, WHF, w2p, K2P, CD2, uc2_b, c0in, 0, C0P);

    // 6) conv stack at full res: M = B*H*W = 17920*64
    const int MB = (int)(BHW / 64);
    k_conv3_wmma<<<dim3(MB), 128, 0, stream>>>(c0in, 5, K0P, c0wp, c0_b, 16,
                                               (const _Float16*)nullptr, h1, CR, nullptr, EP_RELU);
    for (int r = 0; r < 3; ++r) {
        k_conv3_wmma<<<dim3(MB), 128, 0, stream>>>(h1, 4, KRP, rwp[2 * r], rb_[2 * r], 16,
                                                   (const _Float16*)nullptr, h2, CR, nullptr, EP_RELU);
        k_conv3_wmma<<<dim3(MB), 128, 0, stream>>>(h2, 4, KRP, rwp[2 * r + 1], rb_[2 * r + 1], 16,
                                                   h1, h1, CR, nullptr, EP_RESID);
    }
    k_conv3_wmma<<<dim3(MB), 128, 0, stream>>>(h1, 4, KRP, fwp, f_b, 1,
                                               (const _Float16*)nullptr, h2, CR, out, EP_SIGMOID);
}
